// PhysicsInformedNN_88811333746731
// MI455X (gfx1250) — matrix operand order repaired and checked by reading, not compiled, not run on hardware
//
#include <hip/hip_runtime.h>

typedef __attribute__((ext_vector_type(16))) _Float16 v16h;
typedef __attribute__((ext_vector_type(8)))  _Float16 v8h;
typedef __attribute__((ext_vector_type(8)))  float    v8f;

typedef __attribute__((address_space(3))) _Float16 lds_half;
typedef __attribute__((address_space(3))) float    lds_float;
typedef __attribute__((address_space(3))) v8h      lds_v8h;

#define NWAVES 8
#define BLOCK  256
#define MTILE  32
#define LDK    512            // row stride (halves) of activation LDS buffers
#define SBUF   (MTILE * LDK)  // halves per derivative stream
#define DT     0.8f

// Branch-free tanh: hardware v_tanh_f32 if clang exposes it on gfx1250,
// else exp2/rcp identity (saturates correctly: exp2->{0,inf} => rcp fine).
__device__ __forceinline__ float fast_tanh(float x) {
#if __has_builtin(__builtin_amdgcn_tanhf)
    return __builtin_amdgcn_tanhf(x);
#else
    const float e = __builtin_amdgcn_exp2f(x * 2.8853900817779268f); // e^(2x)
    return 1.0f - 2.0f * __builtin_amdgcn_rcpf(e + 1.0f);
#endif
}

__device__ __forceinline__ v8f wmma_f16(v16h a, v16h b, v8f c) {
    return __builtin_amdgcn_wmma_f32_16x16x32_f16(
        false, a, false, b, (short)0, c, false, false);
}

// A-fragment (16x32 f16) from LDS row-major [m][k]: per ISA layout each lane
// needs two contiguous 8-half runs: K = kbase+8*hi..+7 and kbase+16+8*hi..+7.
__device__ __forceinline__ v16h load_A_lds(const lds_half* buf, int m_base,
                                           int k_base, int ldk) {
    const int lane = threadIdx.x & 31;
    const int hi   = lane >> 4;
    const int base = (m_base + (lane & 15)) * ldk + k_base + 8 * hi;
    v8h lo = *(const lds_v8h*)(buf + base);
    v8h hv = *(const lds_v8h*)(buf + base + 16);
    return __builtin_shufflevector(lo, hv, 0, 1, 2, 3, 4, 5, 6, 7,
                                   8, 9, 10, 11, 12, 13, 14, 15);
}

// B-fragment from pre-packed global buffer: fragment-major, lane-major,
// 16 contiguous halves per lane -> two unconditional 16B global loads.
__device__ __forceinline__ v16h load_B_pk(const _Float16* P, int frag) {
    const int lane = threadIdx.x & 31;
    const _Float16* p = P + (((long)frag * 32 + lane) << 4);
    v8h lo = *(const v8h*)p;
    v8h hv = *(const v8h*)(p + 8);
    return __builtin_shufflevector(lo, hv, 0, 1, 2, 3, 4, 5, 6, 7,
                                   8, 9, 10, 11, 12, 13, 14, 15);
}

// Pack W (fout x fin, f32 row-major) into WMMA-B fragment order, f16, zero-pad.
// B[k][n] = W[n][k]; N=lane%16, K=8*(lane/16)+(e&7)+16*(e>>3).
__global__ __launch_bounds__(256) void pack_B(const float* __restrict__ W,
                                              _Float16* __restrict__ dst,
                                              int fin, int fout, int Kpad,
                                              int ntiles) {
    const int total = (Kpad >> 5) * ntiles * 512;
    for (int idx = blockIdx.x * 256 + threadIdx.x; idx < total;
         idx += gridDim.x * 256) {
        const int frag = idx >> 9;
        const int r    = idx & 511;
        const int lane = r >> 4, e = r & 15;
        const int kb   = frag / ntiles, nt = frag % ntiles;
        const int n = nt * 16 + (lane & 15);
        const int k = kb * 32 + 8 * (lane >> 4) + (e & 7) + 16 * (e >> 3);
        const float v = (k < fin && n < fout) ? W[n * fin + k] : 0.0f;
        dst[idx] = (_Float16)v;
    }
}

__global__ __launch_bounds__(256) void pack_bias(const float* __restrict__ b,
                                                 float* __restrict__ dst,
                                                 int fout, int Npad) {
    const int i = blockIdx.x * 256 + threadIdx.x;
    if (i < Npad) dst[i] = (i < fout) ? b[i] : 0.0f;
}

// Hidden layer: (Z,Zx,Zxx) = (H,Hx,Hxx) @ W^T (+b on Z), tanh derivative chain.
template <int KPAD, int NTILES>
__device__ __forceinline__ void layer_gemm_tanh(const lds_half* inb,
                                                lds_half* outb,
                                                const _Float16* P,
                                                const float* bp) {
    const int wave = threadIdx.x >> 5;
    const int lane = threadIdx.x & 31;
    const int hi   = lane >> 4;
    const int nloc = lane & 15;
    for (int task = wave; task < 2 * NTILES; task += NWAVES) {
        const int mt = task & 1;
        const int nt = task >> 1;
        v8f cz = {}, cx = {}, cxx = {};
#pragma unroll
        for (int kb = 0; kb < KPAD; kb += 32) {
            v16h bf = load_B_pk(P, (kb >> 5) * NTILES + nt);
            cz  = wmma_f16(load_A_lds(inb + 0 * SBUF, mt * 16, kb, LDK), bf, cz);
            cx  = wmma_f16(load_A_lds(inb + 1 * SBUF, mt * 16, kb, LDK), bf, cx);
            cxx = wmma_f16(load_A_lds(inb + 2 * SBUF, mt * 16, kb, LDK), bf, cxx);
        }
        const int n = nt * 16 + nloc;
        const float bval = bp[n];  // pre-padded, unconditional
#pragma unroll
        for (int r = 0; r < 8; ++r) {
            const int m = mt * 16 + r + 8 * hi;  // D layout: M = r + 8*(lane>=16)
            const float z = cz[r] + bval;
            const float a = fast_tanh(z);
            const float s = 1.0f - a * a;
            const float ax  = s * cx[r];
            const float axx = s * cxx[r] - 2.0f * a * ax * cx[r];
            outb[0 * SBUF + m * LDK + n] = (_Float16)a;   // tanh(0)=0 in pad cols
            outb[1 * SBUF + m * LDK + n] = (_Float16)ax;  // (B pad cols are 0,
            outb[2 * SBUF + m * LDK + n] = (_Float16)axx; //  bias pad is 0)
        }
    }
}

__global__ __launch_bounds__(BLOCK) void pinn_irk_fused(
    const float* W0, const float* b0,
    const _Float16* P1, const _Float16* P2, const _Float16* P3,
    const _Float16* P4, const _Float16* P5, const _Float16* PA,
    const float* bp1, const float* bp2, const float* bp3,
    const float* bp4, const float* bp5,
    const float* x, const float* bvec, float* out, int Ntot) {
    extern __shared__ __align__(16) char smem_raw[];
    lds_half*  bufA = (lds_half*)(smem_raw);            // 3*32*512 f16 = 98304 B
    lds_half*  bufB = (lds_half*)(smem_raw + 98304);    // 98304 B
    lds_float* Ulds = (lds_float*)(smem_raw + 196608);  // 32*112 f32 = 14336 B
    lds_float* Ff32 = (lds_float*)(smem_raw + 210944);  // 14336 B
    lds_half*  Ff16 = (lds_half*)(smem_raw + 225280);   // 32*128 f16 = 8192 B
    lds_float* fb   = (lds_float*)(smem_raw + 233472);  // 128 B
    lds_float* xlds = (lds_float*)(smem_raw + 233600);  // 128 B  (total 233728)

    const int wg   = blockIdx.x;
    const int tid  = threadIdx.x;
    const int wave = tid >> 5;
    const int lane = tid & 31;
    const int hi   = lane >> 4;
    const int nloc = lane & 15;

    if (tid < MTILE) xlds[tid] = x[wg * MTILE + tid];
    __syncthreads();

    // layer 0: 1 -> 20 (pad 32): h=x, h'=1, h''=0 -> z=W*x+b, z'=W, z''=0
    for (int idx = tid; idx < MTILE * 32; idx += BLOCK) {
        const int m = idx >> 5, n = idx & 31;
        float a = 0.0f, ax = 0.0f, axx = 0.0f;
        if (n < 20) {
            const float w = W0[n];
            const float z = w * xlds[m] + b0[n];
            a = fast_tanh(z);
            const float s = 1.0f - a * a;
            ax  = s * w;
            axx = -2.0f * a * ax * w;
        }
        bufA[0 * SBUF + m * LDK + n] = (_Float16)a;
        bufA[1 * SBUF + m * LDK + n] = (_Float16)ax;
        bufA[2 * SBUF + m * LDK + n] = (_Float16)axx;
    }
    __syncthreads();

    layer_gemm_tanh< 32,  4>(bufA, bufB, P1, bp1); __syncthreads();
    layer_gemm_tanh< 64, 14>(bufB, bufA, P2, bp2); __syncthreads();
    layer_gemm_tanh<224, 32>(bufA, bufB, P3, bp3); __syncthreads();
    layer_gemm_tanh<512, 14>(bufB, bufA, P4, bp4); __syncthreads();

    // layer 5: 200 -> 100 (linear) + output transform + reaction term F
    {
        const int NT5 = 8;  // Q=100 padded to 128 (K of the final GEMM)
        for (int task = wave; task < 2 * NT5; task += NWAVES) {
            const int mt = task & 1, nt = task >> 1;
            v8f cz = {}, cx = {}, cxx = {};
#pragma unroll
            for (int kb = 0; kb < 224; kb += 32) {
                v16h bf = load_B_pk(P5, (kb >> 5) * NT5 + nt);
                cz  = wmma_f16(load_A_lds(bufA + 0 * SBUF, mt * 16, kb, LDK), bf, cz);
                cx  = wmma_f16(load_A_lds(bufA + 1 * SBUF, mt * 16, kb, LDK), bf, cx);
                cxx = wmma_f16(load_A_lds(bufA + 2 * SBUF, mt * 16, kb, LDK), bf, cxx);
            }
            const int n = nt * 16 + nloc;
            const float bval = bp5[n];
#pragma unroll
            for (int r = 0; r < 8; ++r) {
                const int m = mt * 16 + r + 8 * hi;
                const float o = cz[r] + bval, ox = cx[r], oxx = cxx[r];
                const float xv = xlds[m];
                const float g  = xv * xv - 1.0f;
                float u = -1.0f + g * o;
                const float uxx = 2.0f * o + 4.0f * xv * ox + g * oxx;
                float f = -(5.0f * u - 5.0f * u * u * u + 0.0005f * uxx);
                if (n >= 100) { u = 0.0f; f = 0.0f; }
                Ff16[m * 128 + n] = (_Float16)f;
                if (n < 112) { Ulds[m * 112 + n] = u; Ff32[m * 112 + n] = f; }
            }
        }
    }
    __syncthreads();

    // fb[m] = sum_q F[m,q]*bvec[q]   (the (N,1) broadcast term of U1)
    if (tid < MTILE) {
        float acc = 0.0f;
        for (int q = 0; q < 100; ++q) acc += Ff32[tid * 112 + q] * bvec[q];
        fb[tid] = acc;
    }
    __syncthreads();

    // final: G = F @ A^T ; U0 = U + DT*G ; U1 = U0 - DT*fb
    {
        const int NTF = 7;  // 112 cols cover Q=100
        for (int task = wave; task < 2 * NTF; task += NWAVES) {
            const int mt = task & 1, nt = task >> 1;
            v8f c = {};
#pragma unroll
            for (int kb = 0; kb < 128; kb += 32) {
                v16h bf = load_B_pk(PA, (kb >> 5) * NTF + nt);
                c = wmma_f16(load_A_lds(Ff16, mt * 16, kb, 128), bf, c);
            }
            const int q = nt * 16 + nloc;
#pragma unroll
            for (int r = 0; r < 8; ++r) {
                const int m = mt * 16 + r + 8 * hi;
                if (q < 100) {
                    const float u0 = Ulds[m * 112 + q] + DT * c[r];
                    const float u1 = u0 - DT * fb[m];
                    const long  mg = (long)(wg * MTILE + m);
                    out[mg * 100 + q]                    = u0;
                    out[(long)Ntot * 100 + mg * 100 + q] = u1;
                }
            }
        }
    }
}

extern "C" void kernel_launch(void* const* d_in, const int* in_sizes, int n_in,
                              void* d_out, int out_size, void* d_ws, size_t ws_size,
                              hipStream_t stream) {
    const float* W0 = (const float*)d_in[0];
    const float* b0 = (const float*)d_in[1];
    const float* W1 = (const float*)d_in[2];
    const float* b1 = (const float*)d_in[3];
    const float* W2 = (const float*)d_in[4];
    const float* b2 = (const float*)d_in[5];
    const float* W3 = (const float*)d_in[6];
    const float* b3 = (const float*)d_in[7];
    const float* W4 = (const float*)d_in[8];
    const float* b4 = (const float*)d_in[9];
    const float* W5 = (const float*)d_in[10];
    const float* b5 = (const float*)d_in[11];
    const float* x    = (const float*)d_in[12];
    const float* Airk = (const float*)d_in[13];
    const float* bvec = (const float*)d_in[14];
    float* out = (float*)d_out;
    const int N = in_sizes[12];  // 65536

    // Packed-weight layout in d_ws (halves), then padded biases (floats).
    _Float16* wsh = (_Float16*)d_ws;
    _Float16* P1 = wsh + 0;       //  32x64   =   2048
    _Float16* P2 = wsh + 2048;    //  64x224  =  14336
    _Float16* P3 = wsh + 16384;   // 224x512  = 114688
    _Float16* P4 = wsh + 131072;  // 512x224  = 114688
    _Float16* P5 = wsh + 245760;  // 224x128  =  28672
    _Float16* PA = wsh + 274432;  // 128x112  =  14336  (end 288768 h = 577536 B)
    float* wsf = (float*)((char*)d_ws + 577536);
    float* bp1 = wsf + 0;    // 64
    float* bp2 = wsf + 64;   // 224
    float* bp3 = wsf + 288;  // 512
    float* bp4 = wsf + 800;  // 224
    float* bp5 = wsf + 1024; // 128

    auto blocks = [](int total) { return (total + 255) / 256; };
    pack_B<<<blocks((32 / 32) * 4 * 512),  256, 0, stream>>>(W1, P1,  20,  50,  32,  4);
    pack_B<<<blocks((64 / 32) * 14 * 512), 256, 0, stream>>>(W2, P2,  50, 200,  64, 14);
    pack_B<<<blocks((224 / 32) * 32 * 512),256, 0, stream>>>(W3, P3, 200, 500, 224, 32);
    pack_B<<<blocks((512 / 32) * 14 * 512),256, 0, stream>>>(W4, P4, 500, 200, 512, 14);
    pack_B<<<blocks((224 / 32) * 8 * 512), 256, 0, stream>>>(W5, P5, 200, 100, 224,  8);
    pack_B<<<blocks((128 / 32) * 7 * 512), 256, 0, stream>>>(Airk, PA, 100, 100, 128, 7);
    pack_bias<<<1, 256, 0, stream>>>(b1, bp1,  50,  64);
    pack_bias<<<1, 256, 0, stream>>>(b2, bp2, 200, 224);
    pack_bias<<<2, 256, 0, stream>>>(b3, bp3, 500, 512);
    pack_bias<<<1, 256, 0, stream>>>(b4, bp4, 200, 224);
    pack_bias<<<1, 256, 0, stream>>>(b5, bp5, 100, 128);

    const size_t shmem = 233728;  // fits in 320 KB LDS per WGP
    pinn_irk_fused<<<dim3(N / MTILE), dim3(BLOCK), shmem, stream>>>(
        W0, b0, P1, P2, P3, P4, P5, PA, bp1, bp2, bp3, bp4, bp5,
        x, bvec, out, N);
}
